// ModularNetController_86363202388558
// MI455X (gfx1250) — matrix operand order repaired and checked by reading, not compiled, not run on hardware
//
#include <hip/hip_runtime.h>

typedef __attribute__((ext_vector_type(8)))  __bf16 v8bf;
typedef __attribute__((ext_vector_type(16))) __bf16 v16bf;
typedef __attribute__((ext_vector_type(8)))  float  v8f;

constexpr int kB    = 32;
constexpr int kCin  = 128;
constexpr int kCout = 128;
constexpr int kH    = 56;
constexpr int kW    = 56;
constexpr int kE    = 8;
constexpr int kHW   = kH * kW;        // 3136

// ---------------------------------------------------------------------------
// Stage 1: per-(batch, channel) spatial sums.  grid = 32*128 blocks x 256 thr
// ---------------------------------------------------------------------------
__global__ __launch_bounds__(256)
void chan_sum_kernel(const float* __restrict__ x, float* __restrict__ s) {
  const int bc = blockIdx.x;                       // b*128 + c
  const float* xp = x + (size_t)bc * kHW;
  float p = 0.f;
  for (int i = threadIdx.x; i < kHW; i += 256) p += xp[i];
#pragma unroll
  for (int off = 16; off > 0; off >>= 1) p += __shfl_down(p, off, 32);
  __shared__ float red[8];
  const int lane = threadIdx.x & 31, wv = threadIdx.x >> 5;
  if (lane == 0) red[wv] = p;
  __syncthreads();
  if (threadIdx.x == 0) {
    float t = 0.f;
#pragma unroll
    for (int i = 0; i < 8; ++i) t += red[i];
    s[bc] = t;
  }
}

// ---------------------------------------------------------------------------
// Stage 2: logits + argmax routing.  one block, 256 threads = 32 batches x 8E
// ---------------------------------------------------------------------------
__global__ __launch_bounds__(256)
void route_kernel(const float* __restrict__ s, const float* __restrict__ ctlw,
                  const float* __restrict__ ctlb, int* __restrict__ dec) {
  __shared__ float lg[kB][kE];
  const int tid = threadIdx.x;
  const int b = tid >> 3, e = tid & 7;
  float acc = 0.f;
  for (int c = 0; c < kCin; ++c) acc += s[b * kCin + c] * ctlw[e * kCin + c];
  lg[b][e] = acc * (1.0f / (float)kHW) + ctlb[e];
  __syncthreads();
  if (tid < kB) {
    int best = 0;
    float bv = lg[tid][0];
#pragma unroll
    for (int e2 = 1; e2 < kE; ++e2) {
      float v = lg[tid][e2];
      if (v > bv) { bv = v; best = e2; }   // strict > : first-max like argmax
    }
    dec[tid] = best;
  }
}

// ---------------------------------------------------------------------------
// Convert x (f32, [b][c][h][w]) -> bf16 same layout
// ---------------------------------------------------------------------------
__global__ __launch_bounds__(256)
void cvt_x_kernel(const float* __restrict__ x, unsigned short* __restrict__ xb, int n) {
  const int i = blockIdx.x * 256 + threadIdx.x;
  if (i < n) {
    __bf16 h = (__bf16)x[i];
    xb[i] = __builtin_bit_cast(unsigned short, h);
  }
}

// ---------------------------------------------------------------------------
// Convert exp_w (f32, [e][co][ci][3][3]) -> bf16 in WMMA-B-fragment layout:
//   [e][t][ck][co][ci32]   (t = kh*3+kw, ck = ci/32, ci32 = ci%32)
// A wave's 32x16 B-fragment (one 16-co group) is then a fully contiguous
// 1 KB slab: lane L<16 reads [co=cb+L][ci 0..15] (32 B), lane L+16 reads
// [co=cb+L][ci 16..31] — no LDS transpose needed in the GEMM.
// ---------------------------------------------------------------------------
__global__ __launch_bounds__(256)
void cvt_w_kernel(const float* __restrict__ w, unsigned short* __restrict__ wb) {
  const int o = blockIdx.x * 256 + threadIdx.x;
  if (o >= kE * 9 * kCin * kCout) return;
  const int ci = o & 31;
  const int co = (o >> 5) & 127;
  const int ck = (o >> 12) & 3;
  const int t  = (o >> 14) % 9;
  const int e  = o / (9 * kCin * kCout);
  const int in = (((e * kCout + co) * kCin + (ck * 32 + ci)) * 9) + t;
  __bf16 h = (__bf16)w[in];
  wb[o] = __builtin_bit_cast(unsigned short, h);
}

// ---------------------------------------------------------------------------
// Implicit-GEMM routed 3x3 conv via v_wmma_f32_16x16x32_bf16.
// Block = 128 threads (4 waves), one batch x one 8x8 spatial tile (M=64),
// N = 128 Cout.  Wave (m2, n2) owns a 32M x 64N tile = 8 f32 accumulators.
// K = 9 taps x 128 ci, stepped 32 ci: ci-chunk outer (x halo staged once in
// LDS, reused for all 9 taps), tap loop fully unrolled.
// Per K-step per wave: 4 ds_load_b128 (A) + 4 contiguous global v16bf (B)
// feed 8 back-to-back WMMAs; B comes straight from L2 in fragment layout.
// ---------------------------------------------------------------------------
__global__ __launch_bounds__(128)
void moe_conv_kernel(const unsigned short* __restrict__ xb16,
                     const unsigned short* __restrict__ wb16,
                     const float* __restrict__ expb,
                     const int* __restrict__ dec,
                     float* __restrict__ out) {
  __shared__ __attribute__((aligned(32))) __bf16 xlds[100 * 32];   // [pos10x10][ci32]
  __shared__ float outlds[64 * 130];                               // [m][co pad130]

  const int blk   = blockIdx.x;
  const int b     = blk / 49;
  const int tile  = blk % 49;
  const int hbase = (tile / 7) * 8;
  const int wbase = (tile % 7) * 8;
  const int e     = dec[b];

  const int tid  = threadIdx.x;
  const int lane = tid & 31;
  const int wv   = tid >> 5;
  const int m2   = wv & 1;        // M half  (rows m2*32 .. +31)
  const int n2   = wv >> 1;       // N half  (co  n2*64 .. +63)

  const __bf16* xg  = (const __bf16*)xb16 + (size_t)b * kCin * kHW;
  const __bf16* wge = (const __bf16*)wb16 + (size_t)e * 9 * kCin * kCout;

  v8f acc[2][4];
#pragma unroll
  for (int s = 0; s < 2; ++s)
#pragma unroll
    for (int f = 0; f < 4; ++f) { v8f z = {}; acc[s][f] = z; }

  // fixed per-lane fragment geometry
  const int l15   = lane & 15;
  const int mrow0 = m2 * 32 + l15;                 // first M subtile row
  const int hh0   = mrow0 >> 3, ww0 = mrow0 & 7;   // second subtile: +2 rows
  const int ahalf = (lane < 16) ? 0 : 16;          // A: K 0..15 vs 16..31
  const int lane_woff = n2 * 2048 + l15 * 32 + ((lane < 16) ? 0 : 16);

  for (int ck = 0; ck < 4; ++ck) {
    __syncthreads();
    // stage 10x10 halo of 32 channels, transpose to [pos][ci], zero-pad edges
    for (int idx = tid; idx < 3200; idx += 128) {
      const int ci  = idx / 100;
      const int pos = idx - ci * 100;
      const int h   = hbase + pos / 10 - 1;
      const int w   = wbase + pos % 10 - 1;
      float v = 0.f;
      if ((unsigned)h < (unsigned)kH && (unsigned)w < (unsigned)kW)
        v = (float)xg[(ck * 32 + ci) * kHW + h * kW + w];
      xlds[pos * 32 + ci] = (__bf16)v;
    }
    __syncthreads();

#pragma unroll
    for (int t = 0; t < 9; ++t) {
      const int kh = t / 3, kw = t - kh * 3;
      const int q0 = (hh0 + kh) * 10 + (ww0 + kw);   // subtile 0 halo pos
      const int q1 = q0 + 20;                        // subtile 1 = +2 rows

      // B fragments: contiguous, fragment-layout global loads (hit L2)
      const __bf16* wk = wge + (size_t)(t * 4 + ck) * 4096 + lane_woff;
      v16bf bfr[4];
#pragma unroll
      for (int f = 0; f < 4; ++f) bfr[f] = *(const v16bf*)(wk + f * 512);

      // A fragments from LDS halo
      union { v16bf v; v8bf h[2]; } a0, a1;
      a0.h[0] = *(const v8bf*)(xlds + q0 * 32 + ahalf);
      a0.h[1] = *(const v8bf*)(xlds + q0 * 32 + ahalf + 8);
      a1.h[0] = *(const v8bf*)(xlds + q1 * 32 + ahalf);
      a1.h[1] = *(const v8bf*)(xlds + q1 * 32 + ahalf + 8);

      if (t < 8)  // gfx1250 global_prefetch of next tap's weight slab
        __builtin_prefetch((const char*)(wk + 16384) + tid * 64, 0, 1);

#pragma unroll
      for (int f = 0; f < 4; ++f) {
        acc[0][f] = __builtin_amdgcn_wmma_f32_16x16x32_bf16(
            false, a0.v, false, bfr[f], (short)0, acc[0][f], false, false);
        acc[1][f] = __builtin_amdgcn_wmma_f32_16x16x32_bf16(
            false, a1.v, false, bfr[f], (short)0, acc[1][f], false, false);
      }
    }
  }

  // epilogue: accumulators -> LDS (pad stride 130 = bank-conflict-free) -> out
#pragma unroll
  for (int s = 0; s < 2; ++s)
#pragma unroll
    for (int f = 0; f < 4; ++f) {
      const int n  = n2 * 64 + f * 16 + l15;
      const int mb = m2 * 32 + s * 16 + ((lane < 16) ? 0 : 8);
#pragma unroll
      for (int r = 0; r < 8; ++r) outlds[(mb + r) * 130 + n] = acc[s][f][r];
    }
  __syncthreads();
  const float* bias = expb + e * kCout;
  float* ob = out + (size_t)b * kCout * kHW;
  for (int idx = tid; idx < 8192; idx += 128) {
    const int co = idx >> 6;
    const int m  = idx & 63;
    const int oh = hbase + (m >> 3);
    const int ow = wbase + (m & 7);
    ob[co * kHW + oh * kW + ow] = outlds[m * 130 + co] + bias[co];
  }
}

// ---------------------------------------------------------------------------
// Host launcher
// ---------------------------------------------------------------------------
extern "C" void kernel_launch(void* const* d_in, const int* in_sizes, int n_in,
                              void* d_out, int out_size, void* d_ws, size_t ws_size,
                              hipStream_t stream) {
  (void)in_sizes; (void)n_in; (void)out_size; (void)ws_size;
  const float* x    = (const float*)d_in[0];
  const float* ctlw = (const float*)d_in[1];
  const float* ctlb = (const float*)d_in[2];
  const float* expw = (const float*)d_in[3];
  const float* expb = (const float*)d_in[4];
  float* out = (float*)d_out;

  char* ws = (char*)d_ws;
  int*   dec = (int*)ws;                               // 32 ints   (pad 256 B)
  float* s   = (float*)(ws + 256);                     // 4096 f32  (16 KB)
  unsigned short* xb = (unsigned short*)(ws + 16640);  // 25,690,112 B bf16 x
  unsigned short* wb = (unsigned short*)(ws + 16640 + (size_t)kB * kCin * kHW * 2);

  chan_sum_kernel<<<kB * kCin, 256, 0, stream>>>(x, s);
  route_kernel<<<1, 256, 0, stream>>>(s, ctlw, ctlb, dec);

  const int nx = kB * kCin * kHW;
  cvt_x_kernel<<<(nx + 255) / 256, 256, 0, stream>>>(x, xb, nx);
  const int nw = kE * 9 * kCin * kCout;
  cvt_w_kernel<<<(nw + 255) / 256, 256, 0, stream>>>(expw, wb);

  moe_conv_kernel<<<kB * 49, 128, 0, stream>>>(xb, wb, expb, dec, out);
}
